// DynamicViewSampler_62010737819899
// MI455X (gfx1250) — matrix operand order repaired and checked by reading, not compiled, not run on hardware
//
#include <hip/hip_runtime.h>
#include <hip/hip_bf16.h>

typedef __attribute__((ext_vector_type(16))) _Float16 v16h;
typedef __attribute__((ext_vector_type(8)))  _Float16 v8h;
typedef __attribute__((ext_vector_type(4)))  _Float16 v4h;
typedef __attribute__((ext_vector_type(8)))  float    v8f;
typedef __attribute__((ext_vector_type(4)))  float    v4f;

#define GAMMA_F   20.0f
#define LOG2E_F   1.44269504088896f
#define L_DIM     4096
#define D_DIM     1024
#define NVIEWS    64
#define KSTEP     32
#define PITCH     40     // halves per LDS row: 32 data + 8 pad = 80 bytes (16B aligned, bank-spread)
#define FAR_COORD 3e19f  // poison for invalid positions: dist^2 -> inf -> exp2(-inf) = 0

// One workgroup = one (batch, 64-wide N-slab). 512 threads = 16 waves = 4 view-tiles x 4 n-tiles.
__global__ __launch_bounds__(512)
void DynamicViewSampler_62010737819899_kernel(const float* __restrict__ v_pad,
                                              const int*   __restrict__ v_len,
                                              const int*   __restrict__ grid_thws,
                                              const float* __restrict__ centers,
                                              float*       __restrict__ out)
{
    __shared__ float2   Cxy[L_DIM];           // 32KB: per-position coords (validity folded in)
    __shared__ _Float16 Ash[NVIEWS * PITCH];  // m tile, [view][k]
    __shared__ _Float16 Bsh[64 * PITCH];      // v_pad tile transposed, [n][k]
    __shared__ float    Dsh[NVIEWS];          // denominators

    const int b    = blockIdx.y;
    const int d0   = blockIdx.x * 64;
    const int tid  = threadIdx.x;
    const int lane = tid & 31;
    const int wave = tid >> 5;
    const int mi   = wave >> 2;   // view tile 0..3
    const int ni   = wave & 3;    // n tile 0..3

    // ---- per-batch geometry (uniform) ----
    const int   Lv = v_len[b];
    const float Hg = (float)grid_thws[b * 3 + 1];
    const float Wg = (float)grid_thws[b * 3 + 2];
    const float Lf = (float)Lv;
    int W_eff = (int)rintf(sqrtf(Lf * (Wg / Hg)));   // round-half-even like jnp.round
    if (W_eff < 1) W_eff = 1;
    int H_eff = (int)ceilf(Lf / (float)W_eff);
    if (H_eff < 1) H_eff = 1;
    const float invW = 1.0f / (float)W_eff;
    const float invH = 1.0f / (float)H_eff;

    // ---- one-time coordinate table: 8 positions per thread, div/mod out of the hot loop ----
    for (int l = tid; l < L_DIM; l += 512) {
        const int q = l / W_eff;
        const int r = l - q * W_eff;
        float2 c;
        c.x = (float)r * invW;
        c.y = (float)q * invH;
        if (l >= Lv) { c.x = FAR_COORD; c.y = FAR_COORD; }  // exp underflows to exact 0
        Cxy[l] = c;
    }

    // ---- A-producer role: thread computes 4 m-values for a fixed view ----
    const int   av  = tid >> 3;         // view 0..63 (fixed per thread across all k-steps)
    const int   akg = (tid & 7) * 4;    // k offset within 32-step: 0,4,...,28
    const float cx  = centers[((size_t)b * NVIEWS + av) * 2 + 0];
    const float cy  = centers[((size_t)b * NVIEWS + av) * 2 + 1];
    const float escale = -GAMMA_F * LOG2E_F;  // exp(-g*d2) == exp2(escale*d2)

    // ---- B-producer role: thread loads float4 of one v_pad row ----
    const int bk = tid >> 4;            // k row 0..31
    const int bn = (tid & 15) * 4;      // n col 0..60
    const float* bsrc = v_pad + ((size_t)b * L_DIM + bk) * D_DIM + (d0 + bn);

    // ---- WMMA fragment LDS pointers (16B aligned: rows are 80B apart) ----
    const _Float16* arow = &Ash[(mi * 16 + (lane & 15)) * PITCH + ((lane & 16) ? 8 : 0)];
    const _Float16* brow = &Bsh[(ni * 16 + (lane & 15)) * PITCH + ((lane & 16) ? 8 : 0)];

    v8f   acc  = {};
    float dsum = 0.0f;

    __syncthreads();  // coord table ready

    for (int k0 = 0; k0 < L_DIM; k0 += KSTEP) {
        // stage B from global: read-once stream -> non-temporal 128b load
        v4f bv = __builtin_nontemporal_load((const v4f*)bsrc);
        if (k0 + KSTEP < L_DIM)
            __builtin_prefetch(bsrc + (size_t)KSTEP * D_DIM, 0, 3);  // global_prefetch_b8

        // coords for this thread's 4 positions: 2x ds_load_b128
        const float2* cp  = &Cxy[k0 + akg];
        const v4f     c01 = *(const v4f*)(cp);      // x0,y0,x1,y1
        const v4f     c23 = *(const v4f*)(cp + 2);  // x2,y2,x3,y3

        // branchless m computation (validity already folded into coords)
        v4h am;
#pragma unroll
        for (int i = 0; i < 4; ++i) {
            const float x  = (i < 2) ? c01[2 * i]     : c23[2 * i - 4];
            const float y  = (i < 2) ? c01[2 * i + 1] : c23[2 * i - 3];
            const float dx = cx - x;
            const float dy = cy - y;
            const float d2 = fmaf(dx, dx, dy * dy);
            const float e  = __builtin_amdgcn_exp2f(escale * d2);  // v_exp_f32, unconditional
            dsum += e;
            am[i] = (_Float16)e;
        }
        v4h bh;
#pragma unroll
        for (int i = 0; i < 4; ++i) bh[i] = (_Float16)bv[i];

        __syncthreads();  // previous iteration's fragment reads done

        // A tile: [view][k], one 8B store
        *(v4h*)&Ash[av * PITCH + akg] = am;
        // B tile transposed: [n][k], four 2B stores (stride 80B -> spread banks)
        Bsh[(bn + 0) * PITCH + bk] = bh[0];
        Bsh[(bn + 1) * PITCH + bk] = bh[1];
        Bsh[(bn + 2) * PITCH + bk] = bh[2];
        Bsh[(bn + 3) * PITCH + bk] = bh[3];

        __syncthreads();  // tiles visible

        // A 16x32 f16 fragment: lane<16 -> K{0..7 | 16..23}, lane>=16 -> K{8..15 | 24..31}
        v8h alo = *(const v8h*)(arow);
        v8h ahi = *(const v8h*)(arow + 16);
        // B 32x16 f16 fragment: lane<16 -> K{0..7 | 16..23}, lane>=16 -> K{8..15 | 24..31}
        v8h blo = *(const v8h*)(brow);
        v8h bhi = *(const v8h*)(brow + 16);

        v16h A = __builtin_shufflevector(alo, ahi, 0,1,2,3,4,5,6,7,8,9,10,11,12,13,14,15);
        v16h B = __builtin_shufflevector(blo, bhi, 0,1,2,3,4,5,6,7,8,9,10,11,12,13,14,15);

        acc = __builtin_amdgcn_wmma_f32_16x16x32_f16(
                  /*neg_a=*/false, A, /*neg_b=*/false, B,
                  /*c_mod=*/(short)0, acc, /*reuse_a=*/false, /*reuse_b=*/false);

        bsrc += (size_t)KSTEP * D_DIM;
    }

    // ---- denominator: 8 consecutive lanes share a view; reduce within wave ----
    dsum += __shfl_xor(dsum, 1, 32);
    dsum += __shfl_xor(dsum, 2, 32);
    dsum += __shfl_xor(dsum, 4, 32);
    if ((tid & 7) == 0) Dsh[av] = dsum;
    __syncthreads();

    // ---- epilogue: C layout VGPR r -> M = r (lanes 0-15) / 8+r (lanes 16-31), N = lane%16 ----
    const int n_out = d0 + ni * 16 + (lane & 15);
    const int vbase = mi * 16 + ((lane & 16) ? 8 : 0);
#pragma unroll
    for (int r = 0; r < 8; ++r) {
        const int   v  = vbase + r;
        const float dn = Dsh[v] + 1e-06f;
        __builtin_nontemporal_store(acc[r] / dn,
            &out[((size_t)b * NVIEWS + v) * D_DIM + n_out]);
    }
}

extern "C" void kernel_launch(void* const* d_in, const int* in_sizes, int n_in,
                              void* d_out, int out_size, void* d_ws, size_t ws_size,
                              hipStream_t stream) {
    const float* v_pad     = (const float*)d_in[0];
    const int*   v_len     = (const int*)d_in[1];
    const int*   grid_thws = (const int*)d_in[2];
    const float* centers   = (const float*)d_in[3];
    float*       out       = (float*)d_out;

    const int B = in_sizes[1];            // = 16
    dim3 grid(D_DIM / 64, B);             // 16 N-slabs x B batches
    DynamicViewSampler_62010737819899_kernel<<<grid, 512, 0, stream>>>(
        v_pad, v_len, grid_thws, centers, out);
}
